// GraphTransformer_57062935495535
// MI455X (gfx1250) — matrix operand order repaired and checked by reading, not compiled, not run on hardware
//
#include <hip/hip_runtime.h>
#include <hip/hip_bf16.h>

typedef __attribute__((ext_vector_type(16))) _Float16 v16h;
typedef __attribute__((ext_vector_type(8)))  _Float16 v8h;
typedef __attribute__((ext_vector_type(8)))  float    v8f;

#define B_    32
#define NPER  1024
#define NTOT  (B_*NPER)
#define D_    128
#define H_    4
#define HD_   32
#define L_    3
#define DFF_  2048
#define INDIM 64

// ---------------------------------------------------------------------------
// A-fragment loader: 16x32 f16 A matrix per wave (contiguous k order, matched by B).
// lane lo = M row; element i: K = i + 16*hi.
// ---------------------------------------------------------------------------
__device__ __forceinline__ v16h load_a16(const _Float16* __restrict__ A, size_t lda,
                                         int m0, int k0, int lo, int hi) {
  const _Float16* row = A + (size_t)(m0 + lo) * lda + k0;
  v8h p0 = *(const v8h*)(row + 16*hi);
  v8h p1 = *(const v8h*)(row + 16*hi + 8);
  v16h a;
#pragma unroll
  for (int i = 0; i < 8; ++i) { a[i] = p0[i]; a[i+8] = p1[i]; }
  return a;
}

// ---------------------------------------------------------------------------
// Register-blocked WMMA GEMM: C[M,N] = A[M,K](f16) @ W[N,K](f16)^T + bias.
// 256 threads = 8 waves (2 M x 4 N); each wave owns a 32x32 C tile =
// 4 accumulators; per k-step: 2 A-frags + 2 B-frags -> 4 independent WMMAs
// (2x operand reuse + latency hiding on the XDL pipe). Block tile 64x128.
// Optional vt: for the QKV GEMM (N=384), V columns (256..383) are also
// written transposed as vt[b][h][dim][token] so attention P@V gets
// contiguous B-fragments.
// ---------------------------------------------------------------------------
__global__ __launch_bounds__(256) void gemm_wmma(
    const _Float16* __restrict__ A, const _Float16* __restrict__ W,
    const float* __restrict__ bias, float* __restrict__ Cf,
    _Float16* __restrict__ Ch, _Float16* __restrict__ vt,
    int N, int K, int relu)
{
  const int tid  = threadIdx.x;
  const int lane = tid & 31;
  const int wv   = tid >> 5;
  const int lo   = lane & 15, hi = lane >> 4;
  const int m0   = blockIdx.y * 64  + (wv & 1) * 32;
  const int n0   = blockIdx.x * 128 + (wv >> 1) * 32;

  v8f c00 = {}, c01 = {}, c10 = {}, c11 = {};
  for (int k0 = 0; k0 < K; k0 += 32) {
    v16h a0 = load_a16(A, (size_t)K, m0,      k0, lo, hi);
    v16h a1 = load_a16(A, (size_t)K, m0 + 16, k0, lo, hi);
    const _Float16* wr0 = W + (size_t)(n0 + lo) * K      + k0 + 16*hi;
    const _Float16* wr1 = W + (size_t)(n0 + 16 + lo) * K + k0 + 16*hi;
    v16h b0 = *(const v16h*)wr0;
    v16h b1 = *(const v16h*)wr1;
    __builtin_prefetch(wr0 + 32, 0, 1);
    __builtin_prefetch(wr1 + 32, 0, 1);
    c00 = __builtin_amdgcn_wmma_f32_16x16x32_f16(false, a0, false, b0, (short)0, c00, false, false);
    c01 = __builtin_amdgcn_wmma_f32_16x16x32_f16(false, a0, false, b1, (short)0, c01, false, false);
    c10 = __builtin_amdgcn_wmma_f32_16x16x32_f16(false, a1, false, b0, (short)0, c10, false, false);
    c11 = __builtin_amdgcn_wmma_f32_16x16x32_f16(false, a1, false, b1, (short)0, c11, false, false);
  }

  auto store_tile = [&](v8f c, int mt, int nt) {
    const int col = nt + lo;
    const float bv = bias ? bias[col] : 0.f;
#pragma unroll
    for (int r = 0; r < 8; ++r) {
      int row = mt + r + 8*hi;
      float v = c[r] + bv;
      if (relu) v = fmaxf(v, 0.f);
      size_t idx = (size_t)row * N + col;
      if (Cf) Cf[idx] = v;
      if (Ch) Ch[idx] = (_Float16)v;
      if (vt && col >= 2*D_) {           // V columns of the QKV projection
        int hd = col - 2*D_;             // 0..127
        int hh = hd >> 5, dd = hd & 31;  // head, dim-in-head
        int bb = row >> 10, ss = row & (NPER - 1);
        vt[((((size_t)bb * H_ + hh) * HD_ + dd) * NPER) + ss] = (_Float16)v;
      }
    }
  };
  store_tile(c00, m0,      n0);
  store_tile(c01, m0,      n0 + 16);
  store_tile(c10, m0 + 16, n0);
  store_tile(c11, m0 + 16, n0 + 16);
}

// ---------------------------------------------------------------------------
// Attention: one workgroup = (batch b, head h, 16-query block).
// QK^T via WMMA (HD=32 == WMMA K); f16 score row in LDS; 3-pass softmax;
// P@V via WMMA with contiguous B-frags from the transposed V buffer;
// 4 waves split the 1024 keys and merge O with ds_add_f32.
// ---------------------------------------------------------------------------
__global__ __launch_bounds__(128) void attn_wmma(
    const _Float16* __restrict__ qkv, const _Float16* __restrict__ vt,
    _Float16* __restrict__ o)
{
  __shared__ _Float16 sc[16][NPER];   // 32 KB: scores -> probabilities
  __shared__ float red[16][8];
  __shared__ float rowmax[16];
  __shared__ float rowsum[16];
  __shared__ float obuf[16][32];

  const int b  = blockIdx.z, h = blockIdx.y;
  const int q0 = blockIdx.x * 16;
  const int tid  = threadIdx.x;
  const int lane = tid & 31;
  const int wv   = tid >> 5;
  const int lo   = lane & 15, hi = lane >> 4;

  for (int i = tid; i < 16*32; i += 128) ((float*)obuf)[i] = 0.f;

  // Q fragment (A matrix 16x32 f16), loaded once
  const _Float16* qrow = qkv + ((size_t)(b*NPER + q0 + lo) * 384) + h*HD_;
  v8h qa = *(const v8h*)(qrow + 16*hi);
  v8h qb = *(const v8h*)(qrow + 16*hi + 8);
  v16h af;
#pragma unroll
  for (int i = 0; i < 8; ++i) { af[i] = qa[i]; af[i+8] = qb[i]; }

  const float scale = 0.17677669529663687f;  // 1/sqrt(HD)

  // scores: wave wv handles keys [wv*256, wv*256+256)
  for (int kv = wv*256; kv < wv*256 + 256; kv += 16) {
    const _Float16* krow = qkv + ((size_t)(b*NPER + kv + lo) * 384) + D_ + h*HD_ + 16*hi;
    v16h bf = *(const v16h*)krow;  // column of K^T == row of K (contiguous)
    v8f cacc = {};
    cacc = __builtin_amdgcn_wmma_f32_16x16x32_f16(false, af, false, bf,
                                                  (short)0, cacc, false, false);
#pragma unroll
    for (int r = 0; r < 8; ++r)
      sc[r + 8*hi][kv + lo] = (_Float16)(cacc[r] * scale);
  }
  __syncthreads();

  // softmax over 1024 keys, 8 threads per row
  const int rr = tid & 15, ch = tid >> 4;
  {
    float mx = -3.0e38f;
    for (int c = ch*128; c < ch*128 + 128; ++c) mx = fmaxf(mx, (float)sc[rr][c]);
    red[rr][ch] = mx;
  }
  __syncthreads();
  if (tid < 16) {
    float mx = red[tid][0];
#pragma unroll
    for (int j = 1; j < 8; ++j) mx = fmaxf(mx, red[tid][j]);
    rowmax[tid] = mx;
  }
  __syncthreads();
  {
    float s = 0.f, mx = rowmax[rr];
    for (int c = ch*128; c < ch*128 + 128; ++c) {
      float e = __expf((float)sc[rr][c] - mx);
      sc[rr][c] = (_Float16)e;
      s += e;
    }
    red[rr][ch] = s;
  }
  __syncthreads();
  if (tid < 16) {
    float s = 0.f;
#pragma unroll
    for (int j = 0; j < 8; ++j) s += red[tid][j];
    rowsum[tid] = s;
  }
  __syncthreads();
  {
    float inv = 1.f / rowsum[rr];
    for (int c = ch*128; c < ch*128 + 128; ++c)
      sc[rr][c] = (_Float16)((float)sc[rr][c] * inv);
  }
  __syncthreads();

  // P @ V: 32 keys per step; B-frags are contiguous rows of vt[b][h][dim][*]
  const _Float16* vtb = vt + ((size_t)(b*H_ + h) * HD_) * NPER;
  v8f o0 = {}, o1 = {};
  for (int kv = wv*256; kv < wv*256 + 256; kv += 32) {
    const _Float16* prow = &sc[lo][kv];
    v8h p0 = *(const v8h*)(prow + 16*hi);
    v8h p1 = *(const v8h*)(prow + 16*hi + 8);
    v16h pf;
#pragma unroll
    for (int i = 0; i < 8; ++i) { pf[i] = p0[i]; pf[i+8] = p1[i]; }
    v16h b0 = *(const v16h*)(vtb + (size_t)lo        * NPER + kv + 16*hi);
    v16h b1 = *(const v16h*)(vtb + (size_t)(16 + lo) * NPER + kv + 16*hi);
    o0 = __builtin_amdgcn_wmma_f32_16x16x32_f16(false, pf, false, b0, (short)0, o0, false, false);
    o1 = __builtin_amdgcn_wmma_f32_16x16x32_f16(false, pf, false, b1, (short)0, o1, false, false);
  }
#pragma unroll
  for (int r = 0; r < 8; ++r) {
    atomicAdd(&obuf[r + 8*hi][lo],      o0[r]);   // ds_add_f32
    atomicAdd(&obuf[r + 8*hi][16 + lo], o1[r]);
  }
  __syncthreads();
  {
    int i = tid * 4;              // 512 values, 128 threads x 4
    int row = i >> 5, col = i & 31;
#pragma unroll
    for (int j = 0; j < 4; ++j)
      o[((size_t)(b*NPER + q0 + row) * D_) + h*HD_ + col + j] = (_Float16)obuf[row][col + j];
  }
}

// ---------------------------------------------------------------------------
// LayerNorm(base + delta) -> f32 + f16 mirrors. One wave per row of 128.
// ---------------------------------------------------------------------------
__global__ __launch_bounds__(128) void ln_kernel(
    const float* __restrict__ base, const float* __restrict__ delta,
    const float* __restrict__ g, const float* __restrict__ be,
    float* __restrict__ outf, _Float16* __restrict__ outh)
{
  const int row  = blockIdx.x * 4 + (threadIdx.x >> 5);
  const int lane = threadIdx.x & 31;
  size_t off = (size_t)row * D_ + lane * 4;
  float x[4];
#pragma unroll
  for (int i = 0; i < 4; ++i) x[i] = base[off+i] + delta[off+i];
  float s = x[0] + x[1] + x[2] + x[3];
#pragma unroll
  for (int m = 16; m >= 1; m >>= 1) s += __shfl_xor(s, m, 32);
  float mean = s * (1.f / D_);
  float v = 0.f;
#pragma unroll
  for (int i = 0; i < 4; ++i) { float d = x[i] - mean; v += d * d; }
#pragma unroll
  for (int m = 16; m >= 1; m >>= 1) v += __shfl_xor(v, m, 32);
  float inv = rsqrtf(v * (1.f / D_) + 1e-5f);
#pragma unroll
  for (int i = 0; i < 4; ++i) {
    float y = (x[i] - mean) * inv * g[lane*4 + i] + be[lane*4 + i];
    outf[off+i] = y;
    outh[off+i] = (_Float16)y;
  }
}

// mask[row] = (sum_d h[row,d] != 0) — from the input projection, pre-overwrite
__global__ __launch_bounds__(128) void mask_kernel(const float* __restrict__ a,
                                                   float* __restrict__ mask)
{
  const int row  = blockIdx.x * 4 + (threadIdx.x >> 5);
  const int lane = threadIdx.x & 31;
  size_t off = (size_t)row * D_ + lane * 4;
  float s = a[off] + a[off+1] + a[off+2] + a[off+3];
#pragma unroll
  for (int m = 16; m >= 1; m >>= 1) s += __shfl_xor(s, m, 32);
  if (lane == 0) mask[row] = (s != 0.f) ? 1.f : 0.f;
}

__global__ __launch_bounds__(128) void pool_kernel(const float* __restrict__ act,
    const float* __restrict__ mask, float* __restrict__ out)
{
  const int b = blockIdx.x, d = threadIdx.x;
  float a = 0.f, m = 0.f;
  for (int s = 0; s < NPER; ++s) {
    float mk = mask[b*NPER + s];
    a += act[((size_t)(b*NPER + s)) * D_ + d] * mk;
    m += mk;
  }
  out[b*D_ + d] = a / m;
}

__global__ void cvt_kernel(const float* __restrict__ in, _Float16* __restrict__ out, int n)
{
  int i = blockIdx.x * 256 + threadIdx.x;
  if (i < n) out[i] = (_Float16)in[i];
}

// ---------------------------------------------------------------------------
extern "C" void kernel_launch(void* const* d_in, const int* in_sizes, int n_in,
                              void* d_out, int out_size, void* d_ws, size_t ws_size,
                              hipStream_t stream)
{
  (void)in_sizes; (void)n_in; (void)out_size; (void)ws_size;
  const float* x    = (const float*)d_in[0];
  // d_in[1] (batch) is repeat(arange(32),1024): scatter == reshape, unused
  const float* W_in = (const float*)d_in[2];
  const float* b_in = (const float*)d_in[3];
  const float* Wqkv = (const float*)d_in[4];
  const float* bqkv = (const float*)d_in[5];
  const float* Wo   = (const float*)d_in[6];
  const float* bo   = (const float*)d_in[7];
  const float* W1   = (const float*)d_in[8];
  const float* b1   = (const float*)d_in[9];
  const float* W2   = (const float*)d_in[10];
  const float* b2   = (const float*)d_in[11];
  const float* ln1g = (const float*)d_in[12];
  const float* ln1b = (const float*)d_in[13];
  const float* ln2g = (const float*)d_in[14];
  const float* ln2b = (const float*)d_in[15];
  float* out = (float*)d_out;

  char* wsp = (char*)d_ws; size_t off = 0;
  auto alloc = [&](size_t bytes) -> void* {
    void* p = wsp + off;
    off = (off + bytes + 255) & ~(size_t)255;
    return p;
  };
  float*    act   = (float*)   alloc((size_t)NTOT * D_ * 4);
  _Float16* acth  = (_Float16*)alloc((size_t)NTOT * D_ * 2);
  _Float16* xh    = (_Float16*)alloc((size_t)NTOT * INDIM * 2);
  _Float16* winh  = (_Float16*)alloc((size_t)D_ * INDIM * 2);
  _Float16* wqkvh = (_Float16*)alloc((size_t)L_ * 3 * D_ * D_ * 2);
  _Float16* woh   = (_Float16*)alloc((size_t)L_ * D_ * D_ * 2);
  _Float16* w1h   = (_Float16*)alloc((size_t)L_ * DFF_ * D_ * 2);
  _Float16* w2h   = (_Float16*)alloc((size_t)L_ * D_ * DFF_ * 2);
  _Float16* qkvh  = (_Float16*)alloc((size_t)NTOT * 3 * D_ * 2);
  _Float16* vtb   = (_Float16*)alloc((size_t)B_ * H_ * HD_ * NPER * 2);
  _Float16* oh    = (_Float16*)alloc((size_t)NTOT * D_ * 2);
  float*    tmp   = (float*)   alloc((size_t)NTOT * D_ * 4);
  _Float16* midh  = (_Float16*)alloc((size_t)8192 * DFF_ * 2);
  float*    maskb = (float*)   alloc((size_t)NTOT * 4);

  auto cvt = [&](const float* src, _Float16* dst, int n) {
    cvt_kernel<<<(n + 255)/256, 256, 0, stream>>>(src, dst, n);
  };
  cvt(x,    xh,    NTOT * INDIM);
  cvt(W_in, winh,  D_ * INDIM);
  cvt(Wqkv, wqkvh, L_ * 3 * D_ * D_);
  cvt(Wo,   woh,   L_ * D_ * D_);
  cvt(W1,   w1h,   L_ * DFF_ * D_);
  cvt(W2,   w2h,   L_ * D_ * DFF_);

  // input projection: act/acth = x @ W_in^T + b_in  (padded == reshape)
  gemm_wmma<<<dim3(D_/128, NTOT/64), 256, 0, stream>>>(
      xh, winh, b_in, act, acth, nullptr, D_, INDIM, 0);
  mask_kernel<<<NTOT/4, 128, 0, stream>>>(act, maskb);

  for (int l = 0; l < L_; ++l) {
    // QKV projection (f16 out; V also written transposed for attention)
    gemm_wmma<<<dim3(3*D_/128, NTOT/64), 256, 0, stream>>>(
        acth, wqkvh + (size_t)l*3*D_*D_, bqkv + l*3*D_, nullptr, qkvh, vtb,
        3*D_, D_, 0);
    // fused scaled-dot-product attention
    attn_wmma<<<dim3(NPER/16, H_, B_), 128, 0, stream>>>(qkvh, vtb, oh);
    // output projection
    gemm_wmma<<<dim3(D_/128, NTOT/64), 256, 0, stream>>>(
        oh, woh + (size_t)l*D_*D_, bo + l*D_, tmp, nullptr, nullptr,
        D_, D_, 0);
    // LN1(residual)
    ln_kernel<<<NTOT/4, 128, 0, stream>>>(act, tmp, ln1g + l*D_, ln1b + l*D_, act, acth);
    // FFN, chunked over rows to bound the 2048-wide intermediate
    for (int c0 = 0; c0 < NTOT; c0 += 8192) {
      gemm_wmma<<<dim3(DFF_/128, 8192/64), 256, 0, stream>>>(
          acth + (size_t)c0*D_, w1h + (size_t)l*DFF_*D_, b1 + l*DFF_,
          nullptr, midh, nullptr, DFF_, D_, 1);
      gemm_wmma<<<dim3(D_/128, 8192/64), 256, 0, stream>>>(
          midh, w2h + (size_t)l*D_*DFF_, b2 + l*D_,
          tmp + (size_t)c0*D_, nullptr, nullptr, D_, DFF_, 0);
    }
    // LN2(residual)
    ln_kernel<<<NTOT/4, 128, 0, stream>>>(act, tmp, ln2g + l*D_, ln2b + l*D_, act, acth);
  }

  pool_kernel<<<B_, 128, 0, stream>>>(act, maskb, out);
}
